// ContrastiveLoss_85564338471659
// MI455X (gfx1250) — matrix operand order repaired and checked by reading, not compiled, not run on hardware
//
#include <hip/hip_runtime.h>
#include <hip/hip_bf16.h>

typedef __attribute__((ext_vector_type(16))) _Float16 v16h;
typedef __attribute__((ext_vector_type(8)))  _Float16 v8h;
typedef __attribute__((ext_vector_type(4)))  _Float16 v4h;
typedef __attribute__((ext_vector_type(8)))  float    v8f;

#define K_D     1024          // embedding dim (compile-time)
#define INV_T   10.0f         // 1 / TEMPERATURE
#define BLK     128           // block tile (rows == cols)
#define KC      32            // K chunk staged per LDS buffer (1 WMMA K-step)
#define LSTR    (KC + 8)      // padded LDS row stride (halves): conflict-free b128
#define KSTEPS  (K_D / KC)    // 32

// ---------------- kernel 0: zero the row accumulators ----------------
__global__ void CL_zero_kernel(float* p, int n) {
    int i = blockIdx.x * blockDim.x + threadIdx.x;
    if (i < n) p[i] = 0.0f;
}

// ---------------- kernel 1: f32 -> f16 convert ----------------
__global__ void CL_cvt_kernel(const float* __restrict__ in, _Float16* __restrict__ out, int n) {
    int i = (blockIdx.x * blockDim.x + threadIdx.x) * 4;
    if (i + 3 < n) {
        float4 f = *(const float4*)(in + i);
        v4h h;
        h[0] = (_Float16)f.x; h[1] = (_Float16)f.y;
        h[2] = (_Float16)f.z; h[3] = (_Float16)f.w;
        *(v4h*)(out + i) = h;
    } else {
        for (; i < n; ++i) out[i] = (_Float16)in[i];
    }
}

// ---------------- kernel 2: upper-triangular block GEMM + exp + sums ----
// Grid (B,B), B = N/128; block (bi,bj) with bj<bi exits. 256 threads = 8
// waves in a 4x2 grid; each wave computes a 32x64 sub-tile (2x4 WMMA accs).
// Software-pipelined double-buffered LDS staging, ONE barrier per K-step:
// a wave only signals the barrier after s_wait_dscnt 0 drained its previous
// fragment reads, so next-iteration stores to the other buffer cannot race.
__global__ __launch_bounds__(256)
void CL_wmma_kernel(const _Float16* __restrict__ E,
                    const int* __restrict__ labels,
                    float* __restrict__ allS,
                    float* __restrict__ posS,
                    int N) {
    const int bi = blockIdx.x;
    const int bj = blockIdx.y;
    if (bj < bi) return;
    const bool diag = (bi == bj);

    __shared__ _Float16 sA[2][BLK * LSTR];   // 2 x 10 KB
    __shared__ _Float16 sB[2][BLK * LSTR];   // 2 x 10 KB

    const int lane   = threadIdx.x & 31;
    const int wid    = threadIdx.x >> 5;     // 0..7
    const int waveM  = wid & 3;              // 4 row groups of 32
    const int waveN  = wid >> 2;             // 2 col groups of 64
    const int laneLo = lane & 15;
    const int laneHi = lane >> 4;            // 0/1 K-half selector

    const int rBlock = bi * BLK;
    const int cBlock = bj * BLK;
    const int rWave  = rBlock + waveM * 32;  // wave row base (2 tiles of 16)
    const int cWave  = cBlock + waveN * 64;  // wave col base (4 tiles of 16)

    // Per-thread staging decomposition: 1024 v8h chunks per K-step,
    // 4 per thread.  c = tid + it*256 -> mat / row / col.
    int stMat[4], stRow[4], stCol[4];
#pragma unroll
    for (int it = 0; it < 4; ++it) {
        const int c = threadIdx.x + it * 256;
        stMat[it] = c >> 9;                // 0 = A, 1 = B
        stRow[it] = (c >> 2) & (BLK - 1);
        stCol[it] = (c & 3) * 8;
    }

    v8f acc[2][4];
#pragma unroll
    for (int tm = 0; tm < 2; ++tm)
#pragma unroll
        for (int tn = 0; tn < 4; ++tn) acc[tm][tn] = (v8f){};

    // prologue: fetch chunk 0 into registers
    v8h g[4];
#pragma unroll
    for (int it = 0; it < 4; ++it)
        g[it] = *(const v8h*)(E +
            (size_t)((stMat[it] ? cBlock : rBlock) + stRow[it]) * K_D + stCol[it]);

#pragma unroll 2
    for (int kc = 0; kc < KSTEPS; ++kc) {
        const int buf = kc & 1;

        // commit staged chunk kc to LDS buffer `buf`
#pragma unroll
        for (int it = 0; it < 4; ++it) {
            _Float16* dst = (stMat[it] ? sB[buf] : sA[buf]) +
                            stRow[it] * LSTR + stCol[it];
            *(v8h*)dst = g[it];
        }
        // issue global loads for chunk kc+1 (latency hidden by WMMA chain)
        if (kc + 1 < KSTEPS) {
            const int k0 = (kc + 1) * KC;
#pragma unroll
            for (int it = 0; it < 4; ++it)
                g[it] = *(const v8h*)(E +
                    (size_t)((stMat[it] ? cBlock : rBlock) + stRow[it]) * K_D +
                    k0 + stCol[it]);
        }
        __syncthreads();

        // ---- preload ALL fragments, then 8 back-to-back WMMAs ----
        v16h afrag[2];
#pragma unroll
        for (int tm = 0; tm < 2; ++tm) {
            // A 16x32 layout: lanes<16 K{0..7,16..23}, lanes>=16 K{8..15,24..31}
            const _Float16* ap =
                sA[buf] + (waveM * 32 + tm * 16 + laneLo) * LSTR;
            v8h a0 = *(const v8h*)(ap + 8 * laneHi);
            v8h a1 = *(const v8h*)(ap + 16 + 8 * laneHi);
#pragma unroll
            for (int t = 0; t < 8; ++t) {
                afrag[tm][t]     = a0[t];
                afrag[tm][t + 8] = a1[t];
            }
        }
        v16h bfrag[4];
#pragma unroll
        for (int tn = 0; tn < 4; ++tn) {
            // B 32x16 layout: lanes<16 K{0..7,16..23}, lanes>=16 K{8..15,24..31}
            const _Float16* bp =
                sB[buf] + (waveN * 64 + tn * 16 + laneLo) * LSTR + 8 * laneHi;
            v8h b0 = *(const v8h*)bp;
            v8h b1 = *(const v8h*)(bp + 16);
#pragma unroll
            for (int t = 0; t < 8; ++t) { bfrag[tn][t] = b0[t]; bfrag[tn][t + 8] = b1[t]; }
        }
        // keep the fragment loads grouped ahead of the WMMA chain
        __builtin_amdgcn_sched_barrier(0);
#pragma unroll
        for (int tn = 0; tn < 4; ++tn)
#pragma unroll
            for (int tm = 0; tm < 2; ++tm)
                acc[tm][tn] = __builtin_amdgcn_wmma_f32_16x16x32_f16(
                    false, afrag[tm], false, bfrag[tn], (short)0, acc[tm][tn],
                    false, false);
        // NOTE: no trailing barrier -- the next iteration writes the OTHER
        // buffer, and the mid-iteration barrier (with its dscnt drain)
        // already orders reads of this buffer against its next overwrite.
    }

    // ---- epilogue: exp, mask self, row-side + col-side accumulation ----
    float labM[2][8];
#pragma unroll
    for (int tm = 0; tm < 2; ++tm)
#pragma unroll
        for (int v = 0; v < 8; ++v)
            labM[tm][v] = (float)labels[rWave + tm * 16 + v + 8 * laneHi];

    float rowAll[2][8], rowPos[2][8];
#pragma unroll
    for (int tm = 0; tm < 2; ++tm)
#pragma unroll
        for (int v = 0; v < 8; ++v) { rowAll[tm][v] = 0.f; rowPos[tm][v] = 0.f; }
    float colAll[4] = {0.f, 0.f, 0.f, 0.f};
    float colPos[4] = {0.f, 0.f, 0.f, 0.f};

#pragma unroll
    for (int tn = 0; tn < 4; ++tn) {
        const int   gN   = cWave + tn * 16 + laneLo;
        const float labN = (float)labels[gN];
#pragma unroll
        for (int tm = 0; tm < 2; ++tm) {
#pragma unroll
            for (int v = 0; v < 8; ++v) {
                const int   gM = rWave + tm * 16 + v + 8 * laneHi;
                const float e  = __expf(acc[tm][tn][v] * INV_T);
                const float w  = (!diag || (gM != gN)) ? 1.0f : 0.0f;
                rowAll[tm][v] += w * e;
                rowPos[tm][v] += w * labN * e;
                if (!diag) {                    // symmetric credit to row gN
                    colAll[tn] += e;
                    colPos[tn] += labM[tm][v] * e;
                }
            }
        }
    }

    // row-side: reduce over the 16 N-lanes, commit rows of the bi strip
#pragma unroll
    for (int tm = 0; tm < 2; ++tm) {
#pragma unroll
        for (int v = 0; v < 8; ++v) {
            float a = rowAll[tm][v], p = rowPos[tm][v];
            for (int m = 8; m >= 1; m >>= 1) {
                a += __shfl_xor(a, m, 16);
                p += __shfl_xor(p, m, 16);
            }
            if (laneLo == 0) {
                const int gM = rWave + tm * 16 + v + 8 * laneHi;
                atomicAdd(&allS[gM], a);
                atomicAdd(&posS[gM], p);
            }
        }
    }
    // col-side: reduce lane <-> lane+16 pair, commit rows of the bj strip
    if (!diag) {
#pragma unroll
        for (int tn = 0; tn < 4; ++tn) {
            float a = colAll[tn] + __shfl_xor(colAll[tn], 16, 32);
            float p = colPos[tn] + __shfl_xor(colPos[tn], 16, 32);
            if (laneHi == 0) {
                const int gN = cWave + tn * 16 + laneLo;
                atomicAdd(&allS[gN], a);
                atomicAdd(&posS[gN], p);
            }
        }
    }
}

// ---------------- kernel 3: finalize scalar loss ----------------
__global__ void CL_finalize_kernel(const float* __restrict__ allS,
                                   const float* __restrict__ posS,
                                   const int* __restrict__ labels,
                                   float* __restrict__ out, int N) {
    __shared__ float sLoss[256];
    __shared__ int   sCnt[256];
    float loss = 0.0f;
    int cnt = 0;
    for (int i = threadIdx.x; i < N; i += blockDim.x) {
        const int l = labels[i];
        cnt += l;
        if (l > 0) loss += -__logf(posS[i] / (allS[i] + 1e-8f));
    }
    sLoss[threadIdx.x] = loss;
    sCnt[threadIdx.x]  = cnt;
    __syncthreads();
    for (int s = 128; s > 0; s >>= 1) {
        if (threadIdx.x < s) {
            sLoss[threadIdx.x] += sLoss[threadIdx.x + s];
            sCnt[threadIdx.x]  += sCnt[threadIdx.x + s];
        }
        __syncthreads();
    }
    if (threadIdx.x == 0) {
        const int n_ref = sCnt[0];
        out[0] = (n_ref < 2) ? 0.0f : sLoss[0] / (float)n_ref;
    }
}

extern "C" void kernel_launch(void* const* d_in, const int* in_sizes, int n_in,
                              void* d_out, int out_size, void* d_ws, size_t ws_size,
                              hipStream_t stream) {
    const float* emb    = (const float*)d_in[0];
    const int*   labels = (const int*)d_in[1];
    float*       out    = (float*)d_out;

    const int nElem = in_sizes[0];     // N*D
    const int N     = in_sizes[1];     // 8192 (D assumed == K_D == 1024)

    _Float16* embH = (_Float16*)d_ws;
    float*    allS = (float*)((char*)d_ws + (size_t)nElem * sizeof(_Float16));
    float*    posS = allS + N;

    // 0) zero accumulators (every call; deterministic)
    CL_zero_kernel<<<(2 * N + 255) / 256, 256, 0, stream>>>(allS, 2 * N);

    // 1) f32 -> f16
    CL_cvt_kernel<<<(nElem / 4 + 255) / 256, 256, 0, stream>>>(emb, embH, nElem);

    // 2) upper-triangular fused WMMA sim + exp + row sums
    const int B = N / BLK;
    CL_wmma_kernel<<<dim3(B, B), 256, 0, stream>>>(embH, labels, allS, posS, N);

    // 3) finalize
    CL_finalize_kernel<<<1, 256, 0, stream>>>(allS, posS, labels, out, N);
}
